// CrossAttentionBlock_44513041055932
// MI455X (gfx1250) — matrix operand order repaired and checked by reading, not compiled, not run on hardware
//
#include <hip/hip_runtime.h>
#include <hip/hip_bf16.h>

// CrossAttentionBlock for MI455X (gfx1250, wave32, WMMA).
// B=2, C=128, HW=4096, 4 heads x hd=32, 32 groups GroupNorm.
// Compute-bound (~34 GFLOP attention vs ~15MB traffic) -> everything on
// v_wmma_f32_16x16x32_f16; K/V tiles streamed to LDS with the TDM
// (tensor_load_to_lds) double-buffered, waited with s_wait_tensorcnt.

typedef __attribute__((ext_vector_type(16))) _Float16 v16h;
typedef __attribute__((ext_vector_type(8)))  _Float16 v8h;
typedef __attribute__((ext_vector_type(8)))  float    v8f;
typedef __attribute__((ext_vector_type(4)))  unsigned int u32x4;
typedef __attribute__((ext_vector_type(8)))  int      i32x8;
typedef __attribute__((ext_vector_type(4)))  int      i32x4;

#define DEV __device__ __forceinline__

static constexpr int L  = 4096;   // H*W
static constexpr int C  = 128;
static constexpr int B  = 2;
static constexpr int HD = 32;     // head dim
static constexpr int NH = 4;

// ---- fragment loaders (16-bit, contiguous k = 16*(lane/16) + s, same order for A and B) ----
// A (16x32, MxK): lane<16 -> row M=lane, K = 0..15;
//                 lane>=16 -> row M=lane-16, K = 16..31.
DEV v16h load_frag_a(const _Float16* base, int stride, int lane) {
  int r = lane & 15, k0 = (lane >> 4) << 4;
  const _Float16* p = base + (size_t)r * stride + k0;
  union { v16h v; v8h h[2]; } f;
  f.h[0] = *(const v8h*)(p);
  f.h[1] = *(const v8h*)(p + 8);
  return f.v;
}
// B (32x16, KxN): lane<16 -> col N=lane, K=0..15; lane>=16 -> col N=lane-16, K=16..31.
DEV v16h load_frag_b(const _Float16* base, int stride, int lane) {
  int n = lane & 15, k0 = (lane >> 4) << 4;
  const _Float16* p = base + (size_t)n * stride + k0;
  union { v16h v; v8h h[2]; } f;
  f.h[0] = *(const v8h*)(p);
  f.h[1] = *(const v8h*)(p + 8);
  return f.v;
}
DEV v8f wmma_f16(v16h a, v16h b, v8f c) {
  return __builtin_amdgcn_wmma_f32_16x16x32_f16(false, a, false, b, (short)0, c,
                                                false, false);
}

// ---- Tensor Data Mover: 2-D tile load to LDS (D# per ISA §8.3/8.4) ----
DEV void tdm_load_2d(unsigned lds_byte_addr, const _Float16* gptr,
                     unsigned tile_d0, unsigned tile_d1,
                     unsigned tensor_d0, unsigned tensor_d1,
                     unsigned long long stride0_elems) {
  unsigned long long ga = (unsigned long long)(size_t)gptr;
  u32x4 g0;
  g0.x = 1u;                                   // count=1, user descriptor
  g0.y = lds_byte_addr;                        // LDS byte address
  g0.z = (unsigned)ga;                         // global_addr[31:0]
  g0.w = (unsigned)((ga >> 32) & 0x01ffffffu) | (2u << 30);  // addr[56:32]|type=2
  i32x8 g1;
  g1[0] = (int)(1u << 16);                     // data_size=1 (2 bytes)
  g1[1] = (int)((tensor_d0 & 0xffffu) << 16);  // tensor_dim0[15:0] @bits63:48
  g1[2] = (int)(((tensor_d0 >> 16) & 0xffffu) | ((tensor_d1 & 0xffffu) << 16));
  g1[3] = (int)(((tensor_d1 >> 16) & 0xffffu) | ((tile_d0 & 0xffffu) << 16));
  g1[4] = (int)(tile_d1 & 0xffffu);            // tile_dim1 (tile_dim2 = 0)
  g1[5] = (int)(stride0_elems & 0xffffffffu);  // tensor_dim0_stride[31:0]
  g1[6] = (int)((stride0_elems >> 32) & 0xffffu);
  g1[7] = 0;
  i32x4 z4 = {0, 0, 0, 0};                     // groups 2/3 unused (2-D tensor)
  i32x8 z8 = {0, 0, 0, 0, 0, 0, 0, 0};
  __builtin_amdgcn_tensor_load_to_lds(g0, g1, z4, z4, z8, 0);
}

// ============================ stage 1: GN stats =============================
__global__ void gn_stats_kernel(const float* __restrict__ x,
                                const float* __restrict__ ctx,
                                float* __restrict__ stats) {
  int bi = blockIdx.x;                 // 0..127: tensor(2) x batch(2) x group(32)
  int t = bi >> 6, b = (bi >> 5) & 1, g = bi & 31;
  const float* src = (t == 0 ? x : ctx) + ((size_t)b * C + g * 4) * L;
  float s = 0.f, s2 = 0.f;
  for (int i = threadIdx.x; i < 4 * L; i += 256) {
    float v = src[i];
    s += v; s2 += v * v;
  }
  __shared__ float r0[256], r1[256];
  r0[threadIdx.x] = s; r1[threadIdx.x] = s2;
  __syncthreads();
  for (int off = 128; off > 0; off >>= 1) {
    if (threadIdx.x < (unsigned)off) {
      r0[threadIdx.x] += r0[threadIdx.x + off];
      r1[threadIdx.x] += r1[threadIdx.x + off];
    }
    __syncthreads();
  }
  if (threadIdx.x == 0) {
    float mean = r0[0] * (1.f / 16384.f);
    float var  = r1[0] * (1.f / 16384.f) - mean * mean;
    stats[bi * 2 + 0] = mean;
    stats[bi * 2 + 1] = rsqrtf(var + 1e-5f);
  }
}

// ================ stage 2: normalize + transpose + f16 cast =================
// xnT  = GN(x)^T   [b][p][c] f16   (A operand of Q projection)
// cnT  = GN(ctx)^T [b][p][c] f16   (A operand of K projection)
// cT   = ctx^T     [b][p][c] f16   (B operand of V projection)
__global__ void norm_tr_kernel(const float* __restrict__ x,
                               const float* __restrict__ ctx,
                               const float* __restrict__ gq, const float* __restrict__ bq,
                               const float* __restrict__ gc, const float* __restrict__ bc,
                               const float* __restrict__ stats,
                               _Float16* __restrict__ xnT,
                               _Float16* __restrict__ cnT,
                               _Float16* __restrict__ cT) {
  size_t idx = (size_t)blockIdx.x * 256 + threadIdx.x;
  const size_t per = (size_t)B * C * L;
  int sel = (int)(idx / per);
  size_t r = idx % per;
  int b = (int)(r / ((size_t)C * L));
  int rem = (int)(r % ((size_t)C * L));
  int p = rem >> 7;           // channel fastest -> coalesced f16 writes
  int c = rem & 127;
  const float* src = (sel == 0 ? x : ctx);
  float v = src[((size_t)b * C + c) * L + p];
  if (sel < 2) {
    const float* g  = (sel == 0 ? gq : gc);
    const float* bt = (sel == 0 ? bq : bc);
    int sbase = (sel * 64 + b * 32 + (c >> 2)) * 2;
    v = (v - stats[sbase]) * stats[sbase + 1] * g[c] + bt[c];
  }
  _Float16* dst = (sel == 0 ? xnT : (sel == 1 ? cnT : cT));
  dst[((size_t)b * L + p) * C + c] = (_Float16)v;
}

// ======================= stage 0: weights f32 -> f16 ========================
__global__ void wcvt_kernel(const float* __restrict__ Wq, const float* __restrict__ Wk,
                            const float* __restrict__ Wv, const float* __restrict__ Wo,
                            _Float16* __restrict__ dst) {
  int idx = blockIdx.x * 256 + threadIdx.x;     // 4 * 16384
  int sel = idx >> 14, r = idx & 16383;
  const float* s = (sel == 0 ? Wq : sel == 1 ? Wk : sel == 2 ? Wv : Wo);
  dst[idx] = (_Float16)s[r];
}

// ==================== stage 3: WMMA QKV projections =========================
// q/k: yT[p][o] = sum_c xnT[p][c] * W[o][c]   (A=xnT, B=W rows)
// v  : y[o][p]  = sum_c W[o][c]  * cT[p][c]   (A=W, B=cT rows) -> vT[c][e]
__global__ void __launch_bounds__(256)
proj_kernel(const _Float16* __restrict__ xnT, const _Float16* __restrict__ cnT,
            const _Float16* __restrict__ cT,  const _Float16* __restrict__ Wh,
            _Float16* __restrict__ qT, _Float16* __restrict__ kT,
            _Float16* __restrict__ vT) {
  int lane = threadIdx.x & 31;
  int wid  = (int)((blockIdx.x * 256 + threadIdx.x) >> 5);   // 0..12287
  int kind = wid >> 12;                                      // 0=q 1=k 2=v
  int r    = wid & 4095;
  int b    = r >> 11;
  int t    = r & 2047;
  int pt   = t >> 3, ot = t & 7;

  const _Float16* Ab;
  const _Float16* Bb;
  if (kind == 0)      { Ab = xnT + ((size_t)b * L + pt * 16) * C; Bb = Wh + 0 * 16384 + ot * 16 * C; }
  else if (kind == 1) { Ab = cnT + ((size_t)b * L + pt * 16) * C; Bb = Wh + 1 * 16384 + ot * 16 * C; }
  else                { Ab = Wh + 2 * 16384 + ot * 16 * C;        Bb = cT + ((size_t)b * L + pt * 16) * C; }

  v8f acc = {};
#pragma unroll
  for (int cc = 0; cc < 4; ++cc)
    acc = wmma_f16(load_frag_a(Ab + cc * 32, C, lane),
                   load_frag_b(Bb + cc * 32, C, lane), acc);

  if (kind < 2) {            // D[p,o] -> store [p][o], row stride C
    int n  = ot * 16 + (lane & 15);                 // o
    int m0 = pt * 16 + ((lane >> 4) << 3);          // p
    _Float16* out = (kind == 0 ? qT : kT) + (size_t)b * L * C;
#pragma unroll
    for (int i = 0; i < 8; ++i) out[(size_t)(m0 + i) * C + n] = (_Float16)acc[i];
  } else {                   // D[o,p] -> store vT [o][p], row stride L
    int n  = pt * 16 + (lane & 15);                 // p
    int m0 = ot * 16 + ((lane >> 4) << 3);          // o
    _Float16* out = vT + (size_t)b * C * L;
#pragma unroll
    for (int i = 0; i < 8; ++i) out[(size_t)(m0 + i) * L + n] = (_Float16)acc[i];
  }
}

// ================== stage 4: flash attention (TDM + WMMA) ===================
// Grid: (qblock=32, head=4, batch=2); 256 threads = 8 waves, 16 queries/wave.
// Per key block (64): S^T tiles (M=key,N=query) via 4 WMMAs, online softmax
// with a single shfl_xor(16) per reduction, P->B-fragment via lane-pair swap,
// P.V via 4 WMMAs on V tiles staged in LDS by the Tensor Data Mover.
__global__ void __launch_bounds__(256)
attn_kernel(const _Float16* __restrict__ qT, const _Float16* __restrict__ kT,
            const _Float16* __restrict__ vT, _Float16* __restrict__ outT) {
  int lane = threadIdx.x & 31;
  int wv   = threadIdx.x >> 5;          // 0..7
  int qb   = blockIdx.x;                // 0..31
  int h    = blockIdx.y;                // 0..3
  int b    = blockIdx.z;                // 0..1
  int d0   = qb * 128 + wv * 16;

  __shared__ _Float16 kbuf[2][64 * 32];   // [e][c] tiles
  __shared__ _Float16 vbuf[2][32 * 64];   // [c][e] tiles

  const _Float16* qbase = qT + ((size_t)b * L) * C + h * HD;        // [p][128]
  const _Float16* kbase = kT + ((size_t)b * L) * C + h * HD;        // [p][128]
  const _Float16* vbase = vT + ((size_t)b * C + h * HD) * L;        // [c][L]

  // Q as B-matrix (K=c 32, N=d 16), pre-scaled by hd^-0.5
  v16h qf = load_frag_b(qbase + (size_t)d0 * C, C, lane);
  qf = qf * (_Float16)0.17677669529663687f;

  v8f o0 = {}, o1 = {};
  float m_run = -1e30f, l_run = 0.f;

  const int nkb = L / 64;
  const bool loader = (wv == 0);
  if (loader) {
    tdm_load_2d((unsigned)(size_t)&kbuf[0][0], kbase, 32, 64, 32, 64, C);
    tdm_load_2d((unsigned)(size_t)&vbuf[0][0], vbase, 64, 32, 64, 32, L);
  }

  for (int kb = 0; kb < nkb; ++kb) {
    int cur = kb & 1;
    if (loader) {
      if (kb + 1 < nkb) {
        const _Float16* kg = kbase + (size_t)(kb + 1) * 64 * C;
        const _Float16* vg = vbase + (size_t)(kb + 1) * 64;
        tdm_load_2d((unsigned)(size_t)&kbuf[cur ^ 1][0], kg, 32, 64, 32, 64, C);
        tdm_load_2d((unsigned)(size_t)&vbuf[cur ^ 1][0], vg, 64, 32, 64, 32, L);
        __builtin_amdgcn_s_wait_tensorcnt(2);   // current buffer's 2 loads done
      } else {
        __builtin_amdgcn_s_wait_tensorcnt(0);
      }
    }
    __syncthreads();

    // S^T = K^T Q : 4 key tiles of 16
    const _Float16* kl = &kbuf[cur][0];
    v8f st[4];
#pragma unroll
    for (int et = 0; et < 4; ++et) {
      v16h kf = load_frag_a(kl + et * 16 * 32, 32, lane);
      v8f z = {};
      st[et] = wmma_f16(kf, qf, z);
    }

    // online softmax over the key axis (rows of S^T; one lane pair per query)
    float mx = -1e30f;
#pragma unroll
    for (int t = 0; t < 4; ++t)
#pragma unroll
      for (int i = 0; i < 8; ++i) mx = fmaxf(mx, st[t][i]);
    mx = fmaxf(mx, __shfl_xor(mx, 16));
    float m_new = fmaxf(m_run, mx);
    float corr  = __expf(m_run - m_new);
    m_run = m_new;
    float lacc = 0.f;
#pragma unroll
    for (int t = 0; t < 4; ++t)
#pragma unroll
      for (int i = 0; i < 8; ++i) {
        float pv = __expf(st[t][i] - m_new);
        st[t][i] = pv;
        lacc += pv;
      }
    l_run = l_run * corr + lacc + __shfl_xor(lacc, 16);
    o0 *= corr;
    o1 *= corr;

    // P.V: build P B-fragments (K=e chunk of 32, N=d) from S^T C-layout
    const _Float16* vl = &vbuf[cur][0];
    const bool low = (lane < 16);
#pragma unroll
    for (int kc = 0; kc < 2; ++kc) {
      v8f a0 = st[2 * kc], a1 = st[2 * kc + 1];
      v16h pb;
#pragma unroll
      for (int i = 0; i < 8; ++i) {
        float oth0 = __shfl_xor(a0[i], 16);
        float oth1 = __shfl_xor(a1[i], 16);
        pb[i]     = (_Float16)(low ? a0[i] : oth1);
        pb[i + 8] = (_Float16)(low ? oth0 : a1[i]);
      }
      v16h vf0 = load_frag_a(vl + 0 * 16 * 64 + kc * 32, 64, lane);
      o0 = wmma_f16(vf0, pb, o0);
      v16h vf1 = load_frag_a(vl + 1 * 16 * 64 + kc * 32, 64, lane);
      o1 = wmma_f16(vf1, pb, o1);
    }
    __syncthreads();   // all waves done with buf[cur] before loader overwrites
  }

  // normalize and store outT[b][p][128] (f16, position-major for out-proj)
  float inv = 1.f / l_run;
  o0 *= inv;
  o1 *= inv;
  int d  = d0 + (lane & 15);
  int m0 = (lane >> 4) << 3;
  _Float16* ob = outT + ((size_t)b * L + d) * C + h * HD + m0;
  v8h h0, h1;
#pragma unroll
  for (int i = 0; i < 8; ++i) { h0[i] = (_Float16)o0[i]; h1[i] = (_Float16)o1[i]; }
  *(v8h*)(ob)      = h0;
  *(v8h*)(ob + 16) = h1;
}

// ============== stage 5: WMMA output projection + residual ==================
__global__ void __launch_bounds__(256)
final_kernel(const _Float16* __restrict__ outT, const _Float16* __restrict__ Woh,
             const float* __restrict__ x, const float* __restrict__ bout,
             const float* __restrict__ alpha, float* __restrict__ y) {
  int lane = threadIdx.x & 31;
  int wid  = (int)((blockIdx.x * 256 + threadIdx.x) >> 5);   // 0..4095
  int b    = wid >> 11;
  int t    = wid & 2047;
  int pt   = t >> 3, ot = t & 7;

  const _Float16* Ab = outT + ((size_t)b * L + pt * 16) * C;
  const _Float16* Bb = Woh + ot * 16 * C;
  v8f acc = {};
#pragma unroll
  for (int cc = 0; cc < 4; ++cc)
    acc = wmma_f16(load_frag_a(Ab + cc * 32, C, lane),
                   load_frag_b(Bb + cc * 32, C, lane), acc);

  int o2 = ot * 16 + (lane & 15);
  int p0 = pt * 16 + ((lane >> 4) << 3);
  float a  = *alpha;
  float bo = bout[o2];
  const float* xs = x + ((size_t)b * C + o2) * L + p0;
  float*       ys = y + ((size_t)b * C + o2) * L + p0;
#pragma unroll
  for (int i = 0; i < 8; ++i) ys[i] = xs[i] + a * (acc[i] + bo);
}

// ================================ launch ====================================
extern "C" void kernel_launch(void* const* d_in, const int* in_sizes, int n_in,
                              void* d_out, int out_size, void* d_ws, size_t ws_size,
                              hipStream_t stream) {
  (void)in_sizes; (void)n_in; (void)out_size; (void)ws_size;
  const float* x    = (const float*)d_in[0];
  const float* ctx  = (const float*)d_in[1];
  const float* gq   = (const float*)d_in[2];
  const float* bq   = (const float*)d_in[3];
  const float* gc   = (const float*)d_in[4];
  const float* bc   = (const float*)d_in[5];
  const float* Wq   = (const float*)d_in[6];
  const float* Wk   = (const float*)d_in[7];
  const float* Wv   = (const float*)d_in[8];
  const float* Wo   = (const float*)d_in[9];
  const float* bout = (const float*)d_in[10];
  const float* alph = (const float*)d_in[11];
  float* y = (float*)d_out;

  char* ws = (char*)d_ws;
  constexpr size_t ACT = (size_t)B * L * C * sizeof(_Float16);   // 2 MB each
  float*    stats = (float*)ws;                     // 256 floats
  _Float16* Wh    = (_Float16*)(ws + 1024);         // 4 x 128x128 f16 (128 KB)
  _Float16* xnT   = (_Float16*)(ws + 1024 + 131072);
  _Float16* cnT   = (_Float16*)((char*)xnT + ACT);
  _Float16* cT    = (_Float16*)((char*)cnT + ACT);
  _Float16* qT    = (_Float16*)((char*)cT  + ACT);
  _Float16* kT    = (_Float16*)((char*)qT  + ACT);
  _Float16* vT    = (_Float16*)((char*)kT  + ACT);
  _Float16* outT  = (_Float16*)((char*)vT  + ACT);

  wcvt_kernel<<<256, 256, 0, stream>>>(Wq, Wk, Wv, Wo, Wh);
  gn_stats_kernel<<<128, 256, 0, stream>>>(x, ctx, stats);
  norm_tr_kernel<<<12288, 256, 0, stream>>>(x, ctx, gq, bq, gc, bc, stats,
                                            xnT, cnT, cT);
  proj_kernel<<<1536, 256, 0, stream>>>(xnT, cnT, cT, Wh, qT, kT, vT);
  attn_kernel<<<dim3(32, NH, B), 256, 0, stream>>>(qT, kT, vT, outT);
  final_kernel<<<512, 256, 0, stream>>>(outT, Wh + 3 * 16384, x, bout, alph, y);
}
